// Decoder_94489281051
// MI455X (gfx1250) — compile-verified, matrix operand order checked by reading, not run on hardware
//
#include <hip/hip_runtime.h>
#include <math.h>

// ---------------------------------------------------------------------------
// Decoder (Bahdanau attention + GRU + big output projection) for MI455X.
//   * All matmuls use v_wmma_f32_16x16x32_bf16 (fp32 accumulation).
//   * Output projection (94% of FLOPs) hoisted out of the time loop into one
//     large WMMA GEMM over Y = [h_new | x] rows.  W_out is pre-converted to
//     bf16 in workspace when ws_size allows (removes the per-tile fp32->bf16
//     VALU bottleneck and halves B-side bytes); otherwise converted
//     in-register on the fly.  Path choice depends only on ws_size ->
//     deterministic across calls.
//   * Logits GEMM: each wave computes 16x64 (4 accumulators), A-operand
//     loaded once per K-step (40 B/WMMA), grid ordered M-fast so W_out
//     N-strips stay L2-resident across all M-tiles.
//   * Recurrence runs as a persistent kernel: 32 blocks = 2 batch-groups(16)
//     x 16 slices, grid-synced with a monotonic atomic barrier + s_sleep.
// ---------------------------------------------------------------------------

#define T_STEPS 64
#define BATCH   32
#define SRC     128
#define HID     512
#define EMB     256
#define ENCD    1024
#define GRUIN   1280
#define NGATE   1536
#define YLD     1792      // [ h(512) | emb(256) | context(1024) ]
#define VOCAB   32000
#define NBLK    32        // persistent-kernel grid

typedef __attribute__((ext_vector_type(16))) __bf16    v16bf;
typedef __attribute__((ext_vector_type(8)))  float     v8f;
typedef __attribute__((ext_vector_type(4)))  unsigned  uv4;
typedef __attribute__((ext_vector_type(4)))  float     fv4;

union BFV { uv4 q[2]; v16bf v; unsigned short s[16]; };

__device__ __forceinline__ unsigned short f2bf(float f) {
  unsigned u = __float_as_uint(f);
  u += 0x7fffu + ((u >> 16) & 1u);          // round-to-nearest-even
  return (unsigned short)(u >> 16);
}

// One WMMA K-loop: A row-major [16,K] bf16, B' row-major [16(N),K] bf16.
// Per-lane K chunks per the CDNA5 16-bit operand layout:
//   lanes 0-15: K = k..k+7 and k+16..k+23 ; lanes 16-31: +8 on both.
__device__ __forceinline__ v8f wmma_loop(const unsigned short* __restrict__ Ar,
                                         const unsigned short* __restrict__ Br,
                                         int K, int hlf, v8f c) {
#pragma unroll 4
  for (int k = 0; k < K; k += 32) {
    int k1 = k + hlf * 8;
    BFV a, b;
    a.q[0] = *(const uv4*)(Ar + k1);
    a.q[1] = *(const uv4*)(Ar + k1 + 16);
    b.q[0] = *(const uv4*)(Br + k1);
    b.q[1] = *(const uv4*)(Br + k1 + 16);
    c = __builtin_amdgcn_wmma_f32_16x16x32_bf16(false, a.v, false, b.v,
                                                (short)0, c, false, false);
  }
  return c;
}

// Monotonic grid barrier: counter never reset mid-kernel -> race-free.
__device__ __forceinline__ void grid_barrier(unsigned* bar, unsigned target) {
  __syncthreads();
  if (threadIdx.x == 0) {
    __threadfence();
    atomicAdd(bar, 1u);
    while (__hip_atomic_load(bar, __ATOMIC_RELAXED, __HIP_MEMORY_SCOPE_AGENT) < target)
      __builtin_amdgcn_s_sleep(1);
    __threadfence();
  }
  __syncthreads();
}

// ------------------------------- prep kernels ------------------------------

__global__ void cvt_kernel(const float* __restrict__ in,
                           unsigned short* __restrict__ out, int n) {
  int i = blockIdx.x * 256 + threadIdx.x;
  if (i < n) out[i] = f2bf(in[i]);
}

// Wide fp32->bf16: 8 elements/thread (2x b128 load, 1x b128 store of packed).
__global__ void cvt8_kernel(const float* __restrict__ in,
                            unsigned short* __restrict__ out, int n8) {
  int i = blockIdx.x * 256 + threadIdx.x;
  if (i < n8) {
    const fv4* p = (const fv4*)in + (size_t)i * 2;
    fv4 f0 = p[0], f1 = p[1];
    union { uv4 q; unsigned short s[8]; } u;
    u.s[0] = f2bf(f0.x); u.s[1] = f2bf(f0.y); u.s[2] = f2bf(f0.z); u.s[3] = f2bf(f0.w);
    u.s[4] = f2bf(f1.x); u.s[5] = f2bf(f1.y); u.s[6] = f2bf(f1.z); u.s[7] = f2bf(f1.w);
    ((uv4*)out)[i] = u.q;
  }
}

// in [R rows, C cols] fp32 -> out [C rows, R cols] bf16 (B-operand layout)
__global__ void tcvt_kernel(const float* __restrict__ in,
                            unsigned short* __restrict__ out, int R, int C) {
  int i = blockIdx.x * 256 + threadIdx.x;
  if (i < R * C) {
    int c = i / R, r = i % R;
    out[i] = f2bf(in[(size_t)r * C + c]);
  }
}

__global__ void init_kernel(const float* __restrict__ hidden,
                            float* __restrict__ h,
                            unsigned short* __restrict__ hb,
                            unsigned* __restrict__ bar) {
  int i = blockIdx.x * 256 + threadIdx.x;
  if (i == 0) *bar = 0u;                      // reset barrier every launch
  if (i < BATCH * HID) {
    float v = hidden[i];
    h[i] = v;
    hb[i] = f2bf(v);                          // parity-0 buffer
  }
}

// Y[(b*T+t), 512..768) = emb[tgt[b][t]]  (bf16)
__global__ void embed_kernel(const int* __restrict__ tgt,
                             const float* __restrict__ emb,
                             unsigned short* __restrict__ Y) {
  int row = blockIdx.x;                       // = b*T + t
  int tok = tgt[row];
  int e = threadIdx.x;
  Y[(size_t)row * YLD + HID + e] = f2bf(emb[(size_t)tok * EMB + e]);
}

// ------------------------ generic bf16 WMMA GEMM ---------------------------
// C[M,N] fp32 = A[M,K] bf16 (row-major, lda) x B'[N,K] bf16 (row-major, ldb)
// block = 128 thr (4 waves): block tile 16(M) x 64(N); grid=(M/16, N/64).
__global__ __launch_bounds__(128) void gemm_bf16_kernel(
    const unsigned short* __restrict__ A, int lda,
    const unsigned short* __restrict__ B, int ldb,
    float* __restrict__ C, int ldc, int K) {
  int lane = threadIdx.x & 31, wave = threadIdx.x >> 5;
  int lrow = lane & 15, hlf = lane >> 4;
  int m0 = blockIdx.x * 16;
  int n0 = blockIdx.y * 64 + wave * 16;
  const unsigned short* Ar = A + (size_t)(m0 + lrow) * lda;
  const unsigned short* Br = B + (size_t)(n0 + lrow) * ldb;
  v8f c = {0.f, 0.f, 0.f, 0.f, 0.f, 0.f, 0.f, 0.f};
  c = wmma_loop(Ar, Br, K, hlf, c);
  int n = n0 + lrow;
#pragma unroll
  for (int r = 0; r < 8; r++)
    C[(size_t)(m0 + r + hlf * 8) * ldc + n] = c[r];
}

// ----------------- logits GEMM, fast path (bf16 W_out copy) ----------------
// out[m, v] = Y[m,:] . Wb[v,:] + b_out[v]
// Wave tile 16(M) x 64(N): one A load feeds 4 WMMAs (40 B/WMMA per lane).
// grid = (M/16 = 128 fast, N/256 = 125 slow) for L2-resident W_out strips.
__global__ __launch_bounds__(128) void logits_bf16_kernel(
    const unsigned short* __restrict__ Y,
    const unsigned short* __restrict__ Wb,
    const float* __restrict__ bout,
    float* __restrict__ out) {
  int lane = threadIdx.x & 31, wave = threadIdx.x >> 5;
  int lrow = lane & 15, hlf = lane >> 4;
  int m0 = blockIdx.x * 16;
  int n0 = blockIdx.y * 256 + wave * 64;
  const unsigned short* Ar = Y + (size_t)(m0 + lrow) * YLD;
  const unsigned short* Br0 = Wb + (size_t)(n0 + lrow) * YLD;
  v8f acc[4];
#pragma unroll
  for (int tt = 0; tt < 4; tt++)
    acc[tt] = (v8f){0.f, 0.f, 0.f, 0.f, 0.f, 0.f, 0.f, 0.f};
#pragma unroll 2
  for (int k = 0; k < YLD; k += 32) {
    int k1 = k + hlf * 8;
    BFV a;
    a.q[0] = *(const uv4*)(Ar + k1);
    a.q[1] = *(const uv4*)(Ar + k1 + 16);
    __builtin_prefetch(Br0 + k1 + 64, 0, 1);   // stream next B chunk (gfx1250 global_prefetch)
#pragma unroll
    for (int tt = 0; tt < 4; tt++) {
      const unsigned short* Br = Br0 + (size_t)tt * 16 * YLD;
      BFV b;
      b.q[0] = *(const uv4*)(Br + k1);
      b.q[1] = *(const uv4*)(Br + k1 + 16);
      acc[tt] = __builtin_amdgcn_wmma_f32_16x16x32_bf16(false, a.v, false, b.v,
                                                        (short)0, acc[tt], false, false);
    }
  }
#pragma unroll
  for (int tt = 0; tt < 4; tt++) {
    int n = n0 + tt * 16 + lrow;
    float bias = bout[n];
#pragma unroll
    for (int r = 0; r < 8; r++)
      out[(size_t)(m0 + r + hlf * 8) * VOCAB + n] = acc[tt][r] + bias;
  }
}

// ------------- logits GEMM, fallback (fp32 W_out, convert on the fly) ------
__global__ __launch_bounds__(128) void logits_f32_kernel(
    const unsigned short* __restrict__ Y,
    const float* __restrict__ Wout,
    const float* __restrict__ bout,
    float* __restrict__ out) {
  int lane = threadIdx.x & 31, wave = threadIdx.x >> 5;
  int lrow = lane & 15, hlf = lane >> 4;
  int m0 = blockIdx.x * 16;
  int n0 = blockIdx.y * 64 + wave * 16;
  const unsigned short* Ar = Y + (size_t)(m0 + lrow) * YLD;
  const float* Wr = Wout + (size_t)(n0 + lrow) * YLD;
  v8f c = {0.f, 0.f, 0.f, 0.f, 0.f, 0.f, 0.f, 0.f};
#pragma unroll 2
  for (int k = 0; k < YLD; k += 32) {
    int k1 = k + hlf * 8;
    BFV a, b;
    a.q[0] = *(const uv4*)(Ar + k1);
    a.q[1] = *(const uv4*)(Ar + k1 + 16);
    fv4 f0 = *(const fv4*)(Wr + k1);
    fv4 f1 = *(const fv4*)(Wr + k1 + 4);
    fv4 f2 = *(const fv4*)(Wr + k1 + 16);
    fv4 f3 = *(const fv4*)(Wr + k1 + 20);
    b.s[0] = f2bf(f0.x);  b.s[1] = f2bf(f0.y);  b.s[2] = f2bf(f0.z);  b.s[3] = f2bf(f0.w);
    b.s[4] = f2bf(f1.x);  b.s[5] = f2bf(f1.y);  b.s[6] = f2bf(f1.z);  b.s[7] = f2bf(f1.w);
    b.s[8] = f2bf(f2.x);  b.s[9] = f2bf(f2.y);  b.s[10] = f2bf(f2.z); b.s[11] = f2bf(f2.w);
    b.s[12] = f2bf(f3.x); b.s[13] = f2bf(f3.y); b.s[14] = f2bf(f3.z); b.s[15] = f2bf(f3.w);
    c = __builtin_amdgcn_wmma_f32_16x16x32_bf16(false, a.v, false, b.v,
                                                (short)0, c, false, false);
  }
  int n = n0 + lrow;
  float bias = bout[n];
#pragma unroll
  for (int r = 0; r < 8; r++)
    out[(size_t)(m0 + r + hlf * 8) * VOCAB + n] = c[r] + bias;
}

// ----------------------- persistent recurrence kernel ----------------------
// 32 blocks x 256 thr: block = (batch-group bg in {0,1}, slice in 0..15).
// 4 grid barriers per step.  h_bf16 is ping-ponged by t parity so slices can
// read the full previous h while others write the next one.
__global__ __launch_bounds__(256, 1) void recurrence_kernel(
    const float* __restrict__ enc_out,        // [32,128,1024]
    const float* __restrict__ enc_proj,       // [32,128,512]
    const unsigned char* __restrict__ src_mask, // [32,128] bool
    const float* __restrict__ v_att,          // [512]
    const unsigned short* __restrict__ WdecT, // [512(a),512(d)] bf16
    const unsigned short* __restrict__ Wih,   // [1536,1280] bf16
    const unsigned short* __restrict__ Whh,   // [1536,512]  bf16
    const float* __restrict__ b_ih,
    const float* __restrict__ b_hh,
    float* __restrict__ h,                    // [32,512] fp32 (persistent)
    unsigned short* __restrict__ hb,          // 2 x [32,512] bf16 ping-pong
    float* __restrict__ dec_proj,             // [32,512]
    float* __restrict__ scores,               // [32,128]
    unsigned short* __restrict__ Y,           // [2048,1792] bf16
    unsigned* __restrict__ bar) {
  const int bg = blockIdx.x & 1;
  const int slice = blockIdx.x >> 1;
  const int tid = threadIdx.x;
  const int wave = tid >> 5, lane = tid & 31;
  const int lrow = lane & 15, hlf = lane >> 4;

  __shared__ float alpha[16][128];
  __shared__ float smax[16], sden[16];
  __shared__ float gxs[16][96];
  __shared__ float ghs[16][96];
  unsigned epoch = 0;

  for (int t = 0; t < T_STEPS; ++t) {
    const unsigned short* hbr = hb + (size_t)(t & 1) * (BATCH * HID);
    unsigned short* hbw = hb + (size_t)((t + 1) & 1) * (BATCH * HID);

    // -------- Phase A: dec_proj = h @ W_dec (2 N-tiles/block, waves 0..1)
    if (wave < 2) {
      int nt = slice * 2 + wave;
      const unsigned short* Ar = hbr + (size_t)(bg * 16 + lrow) * HID;
      const unsigned short* Br = WdecT + (size_t)(nt * 16 + lrow) * HID;
      v8f c = {0.f, 0.f, 0.f, 0.f, 0.f, 0.f, 0.f, 0.f};
      c = wmma_loop(Ar, Br, HID, hlf, c);
#pragma unroll
      for (int r = 0; r < 8; r++)
        dec_proj[(size_t)(bg * 16 + r + hlf * 8) * HID + nt * 16 + lrow] = c[r];
    }
    grid_barrier(bar, (++epoch) * NBLK);

    // -------- Phase B: scores[b,s] = v . tanh(enc_proj + dec_proj), masked
#pragma unroll 1
    for (int i = 0; i < 16; i++) {
      int p = wave * 16 + i;                  // 128 (b,s) pairs per block
      int lb = p >> 3;
      int b = bg * 16 + lb;
      int s = slice * 8 + (p & 7);
      const float* ep = enc_proj + (size_t)(b * SRC + s) * HID;
      const float* dp = dec_proj + (size_t)b * HID;
      float acc = 0.f;
#pragma unroll 4
      for (int j = 0; j < 16; j++) {
        int a = lane + 32 * j;
        acc += tanhf(ep[a] + dp[a]) * v_att[a];
      }
#pragma unroll
      for (int m = 16; m >= 1; m >>= 1) acc += __shfl_xor(acc, m, 32);
      if (lane == 0)
        scores[b * SRC + s] = src_mask[b * SRC + s] ? acc : -1e9f;
    }
    grid_barrier(bar, (++epoch) * NBLK);

    // -------- Phase C: softmax + context -> Y context columns (bf16)
#pragma unroll 1
    for (int i = 0; i < 2; i++) {             // wave handles 2 batch rows
      int lb = wave * 2 + i;
      int b = bg * 16 + lb;
      float v0 = scores[b * SRC + lane];
      float v1 = scores[b * SRC + lane + 32];
      float v2 = scores[b * SRC + lane + 64];
      float v3 = scores[b * SRC + lane + 96];
      float mx = fmaxf(fmaxf(v0, v1), fmaxf(v2, v3));
#pragma unroll
      for (int m = 16; m >= 1; m >>= 1) mx = fmaxf(mx, __shfl_xor(mx, m, 32));
      float sm = expf(v0 - mx) + expf(v1 - mx) + expf(v2 - mx) + expf(v3 - mx);
#pragma unroll
      for (int m = 16; m >= 1; m >>= 1) sm += __shfl_xor(sm, m, 32);
      if (lane == 0) { smax[lb] = mx; sden[lb] = sm; }
    }
    __syncthreads();
#pragma unroll 1
    for (int i = 0; i < 8; i++) {
      int id = tid + 256 * i;
      int lb = id >> 7, s = id & 127;
      alpha[lb][s] = expf(scores[(bg * 16 + lb) * SRC + s] - smax[lb]) / sden[lb];
    }
    __syncthreads();
#pragma unroll 1
    for (int i = 0; i < 4; i++) {             // context: 16 b x 64 d per block
      int id = tid + 256 * i;
      int lb = id >> 6, dd = id & 63;
      int b = bg * 16 + lb;
      int d = slice * 64 + dd;
      const float* eb = enc_out + (size_t)b * SRC * ENCD + d;
      float acc = 0.f;
#pragma unroll 4
      for (int s = 0; s < SRC; s++) acc += alpha[lb][s] * eb[(size_t)s * ENCD];
      Y[(size_t)(b * T_STEPS + t) * YLD + (HID + EMB) + d] = f2bf(acc);
    }
    grid_barrier(bar, (++epoch) * NBLK);

    // -------- Phase D: gx = x@W_ih^T, gh = h@W_hh^T (slice of 96 gate rows),
    //          then fused gate math + h update.
    if (wave < 6) {                           // gx: 6 N-tiles x 40 K-iters
      int q = wave >> 1, u = wave & 1;
      int rowbase = q * HID + slice * 32 + u * 16;
      int b = bg * 16 + lrow;
      const unsigned short* Ar = Y + (size_t)(b * T_STEPS + t) * YLD + HID;
      const unsigned short* Br = Wih + (size_t)(rowbase + lrow) * GRUIN;
      v8f c = {0.f, 0.f, 0.f, 0.f, 0.f, 0.f, 0.f, 0.f};
      c = wmma_loop(Ar, Br, GRUIN, hlf, c);
#pragma unroll
      for (int r = 0; r < 8; r++) gxs[r + hlf * 8][wave * 16 + lrow] = c[r];
    } else {                                  // gh: 3 N-tiles x 16 K-iters each
      int first = (wave - 6) * 3;
      for (int tt = first; tt < first + 3; ++tt) {
        int q = tt >> 1, u = tt & 1;
        int rowbase = q * HID + slice * 32 + u * 16;
        const unsigned short* Ar = hbr + (size_t)(bg * 16 + lrow) * HID;
        const unsigned short* Br = Whh + (size_t)(rowbase + lrow) * HID;
        v8f c = {0.f, 0.f, 0.f, 0.f, 0.f, 0.f, 0.f, 0.f};
        c = wmma_loop(Ar, Br, HID, hlf, c);
#pragma unroll
        for (int r = 0; r < 8; r++) ghs[r + hlf * 8][tt * 16 + lrow] = c[r];
      }
    }
    __syncthreads();
#pragma unroll 1
    for (int i = 0; i < 2; i++) {             // gates: 16 b x 32 j per block
      int id = tid + 256 * i;
      int lb = id >> 5, jl = id & 31;
      int b = bg * 16 + lb;
      int j = slice * 32 + jl;
      float rr = gxs[lb][jl] + b_ih[j] + ghs[lb][jl] + b_hh[j];
      float zz = gxs[lb][32 + jl] + b_ih[HID + j] + ghs[lb][32 + jl] + b_hh[HID + j];
      rr = 1.f / (1.f + expf(-rr));
      zz = 1.f / (1.f + expf(-zz));
      float nn = tanhf(gxs[lb][64 + jl] + b_ih[2 * HID + j] +
                       rr * (ghs[lb][64 + jl] + b_hh[2 * HID + j]));
      float hold = h[(size_t)b * HID + j];
      float hnew = (1.f - zz) * nn + zz * hold;
      h[(size_t)b * HID + j] = hnew;
      unsigned short hv = f2bf(hnew);
      hbw[(size_t)b * HID + j] = hv;
      Y[(size_t)(b * T_STEPS + t) * YLD + j] = hv;
    }
    grid_barrier(bar, (++epoch) * NBLK);
  }
}

// ------------------------------- launcher ----------------------------------

extern "C" void kernel_launch(void* const* d_in, const int* in_sizes, int n_in,
                              void* d_out, int out_size, void* d_ws, size_t ws_size,
                              hipStream_t stream) {
  (void)in_sizes; (void)n_in; (void)out_size;
  const float* enc_out = (const float*)d_in[0];
  const float* hidden  = (const float*)d_in[1];
  const unsigned char* src_mask = (const unsigned char*)d_in[2];
  const int*   tgt   = (const int*)d_in[3];
  const float* emb   = (const float*)d_in[4];
  const float* W_enc = (const float*)d_in[5];
  const float* W_dec = (const float*)d_in[6];
  const float* v_att = (const float*)d_in[7];
  const float* W_ih  = (const float*)d_in[8];
  const float* W_hh  = (const float*)d_in[9];
  const float* b_ih  = (const float*)d_in[10];
  const float* b_hh  = (const float*)d_in[11];
  const float* W_out = (const float*)d_in[12];
  const float* b_out = (const float*)d_in[13];
  float* out = (float*)d_out;

  // workspace carve (~31 MB base)
  char* w = (char*)d_ws;
  auto carve = [&](size_t bytes) -> char* {
    char* p = w; w += (bytes + 255) & ~(size_t)255; return p;
  };
  unsigned short* Y     = (unsigned short*)carve((size_t)BATCH * T_STEPS * YLD * 2);
  unsigned short* encb  = (unsigned short*)carve((size_t)BATCH * SRC * ENCD * 2);
  float*          encp  = (float*)carve((size_t)BATCH * SRC * HID * 4);
  unsigned short* WencT = (unsigned short*)carve((size_t)HID * ENCD * 2);
  unsigned short* WdecT = (unsigned short*)carve((size_t)HID * HID * 2);
  unsigned short* Wihb  = (unsigned short*)carve((size_t)NGATE * GRUIN * 2);
  unsigned short* Whhb  = (unsigned short*)carve((size_t)NGATE * HID * 2);
  float*          hbuf  = (float*)carve((size_t)BATCH * HID * 4);
  unsigned short* hb    = (unsigned short*)carve((size_t)2 * BATCH * HID * 2);
  float*          decp  = (float*)carve((size_t)BATCH * HID * 4);
  float*          sc    = (float*)carve((size_t)BATCH * SRC * 4);
  unsigned*       bar   = (unsigned*)carve(256);

  // bf16 W_out copy only if workspace allows (+~109 MiB); decision is a pure
  // function of ws_size -> deterministic across calls / graph replays.
  size_t used = (size_t)(w - (char*)d_ws);
  size_t wout_bytes = (size_t)VOCAB * YLD * 2;
  bool pre_convert = (ws_size >= used + wout_bytes + 512);
  unsigned short* Woutb = nullptr;
  if (pre_convert) Woutb = (unsigned short*)carve(wout_bytes);

  // --- prep: bf16 copies / transposes / embedding / h init (+barrier reset)
  cvt_kernel<<<(NGATE * GRUIN + 255) / 256, 256, 0, stream>>>(W_ih, Wihb, NGATE * GRUIN);
  cvt_kernel<<<(NGATE * HID + 255) / 256, 256, 0, stream>>>(W_hh, Whhb, NGATE * HID);
  cvt_kernel<<<(BATCH * SRC * ENCD + 255) / 256, 256, 0, stream>>>(enc_out, encb, BATCH * SRC * ENCD);
  tcvt_kernel<<<(ENCD * HID + 255) / 256, 256, 0, stream>>>(W_enc, WencT, ENCD, HID);
  tcvt_kernel<<<(HID * HID + 255) / 256, 256, 0, stream>>>(W_dec, WdecT, HID, HID);
  init_kernel<<<64, 256, 0, stream>>>(hidden, hbuf, hb, bar);
  embed_kernel<<<BATCH * T_STEPS, EMB, 0, stream>>>(tgt, emb, Y);
  if (pre_convert) {
    int n8 = VOCAB * YLD / 8;                 // 7.168M threads, 8 elems each
    cvt8_kernel<<<(n8 + 255) / 256, 256, 0, stream>>>(W_out, Woutb, n8);
  }

  // --- enc_proj = encoder_output @ W_enc  (loop-invariant, WMMA)
  gemm_bf16_kernel<<<dim3((BATCH * SRC) / 16, HID / 64), 128, 0, stream>>>(
      encb, ENCD, WencT, ENCD, encp, HID, ENCD);

  // --- sequential attention + GRU recurrence (persistent, grid-synced)
  recurrence_kernel<<<NBLK, 256, 0, stream>>>(
      enc_out, encp, src_mask, v_att, WdecT, Wihb, Whhb, b_ih, b_hh,
      hbuf, hb, decp, sc, Y, bar);

  // --- all logits in one big WMMA GEMM (235 GFLOP), writes d_out directly
  if (pre_convert) {
    logits_bf16_kernel<<<dim3((BATCH * T_STEPS) / 16, VOCAB / 256), 128, 0, stream>>>(
        Y, Woutb, b_out, out);
  } else {
    logits_f32_kernel<<<dim3((BATCH * T_STEPS) / 16, VOCAB / 64), 128, 0, stream>>>(
        Y, W_out, b_out, out);
  }
}
